// CrossAttention_17712445129135
// MI455X (gfx1250) — matrix operand order repaired and checked by reading, not compiled, not run on hardware
//
#include <hip/hip_runtime.h>

typedef __attribute__((ext_vector_type(16))) __bf16 v16bf;
typedef __attribute__((ext_vector_type(8)))  __bf16 v8bf;
typedef __attribute__((ext_vector_type(8)))  float  v8f;
typedef __attribute__((ext_vector_type(4)))  unsigned int u32x4;
typedef __attribute__((ext_vector_type(8)))  int i32x8;
typedef __attribute__((ext_vector_type(4)))  int i32x4;

#define CD      64   // embed dim
#define KT      32   // keys per tile (== WMMA K for the PV gemm)
#define WAVES   8
#define BQ      128  // query rows per block (8 waves x 16 rows)
#define KPITCH  72   // bf16 elems, multiple of 8 -> 16B-aligned rows
#define VPITCH  40
#define PPITCH  40

// max-reduce across each 16-lane half (DPP row ops stay within 16 lanes on wave32)
__device__ __forceinline__ float rowmax16(float x) {
    int t;
    t = __builtin_amdgcn_update_dpp(0, __float_as_int(x), 0xB1, 0xF, 0xF, true);  // quad_perm 1,0,3,2
    x = fmaxf(x, __int_as_float(t));
    t = __builtin_amdgcn_update_dpp(0, __float_as_int(x), 0x4E, 0xF, 0xF, true);  // quad_perm 2,3,0,1
    x = fmaxf(x, __int_as_float(t));
    t = __builtin_amdgcn_update_dpp(0, __float_as_int(x), 0x141, 0xF, 0xF, true); // row_half_mirror
    x = fmaxf(x, __int_as_float(t));
    t = __builtin_amdgcn_update_dpp(0, __float_as_int(x), 0x140, 0xF, 0xF, true); // row_mirror
    x = fmaxf(x, __int_as_float(t));
    return x;
}

// TDM: DMA one KT x CD fp32 tile from global memory into LDS (2-D descriptor,
// ISA cdna5 sec.8). OOB rows (beyond rows_avail) read as zero -> tail handled.
// This toolchain exposes the 6-arg builtin: (g0, g1, g2, g3, g4, cpol).
__device__ __forceinline__ void tdm_load_tile(const float* gsrc, unsigned lds_off,
                                              unsigned rows_avail) {
    const unsigned long long ga = (unsigned long long)(size_t)gsrc;
    u32x4 g0;
    g0[0] = 1u;                                            // count=1, user descriptor
    g0[1] = lds_off;                                       // lds_addr (bytes)
    g0[2] = (unsigned)(ga & 0xFFFFFFFFu);                  // global_addr[31:0]
    g0[3] = (unsigned)((ga >> 32) & 0x01FFFFFFu) | (2u << 30); // ga[56:32] | type=2
    i32x8 g1;
    g1[0] = (int)(2u << 16);                               // wg_mask=0, data_size=4B
    g1[1] = (int)((unsigned)CD << 16);                     // tensor_dim0[15:0]=64
    g1[2] = (int)((rows_avail & 0xFFFFu) << 16);           // td0 hi=0 | tensor_dim1 lo
    g1[3] = (int)((rows_avail >> 16) & 0xFFFFu) | (int)((unsigned)CD << 16); // td1 hi | tile_dim0=64
    g1[4] = KT;                                            // tile_dim1=32, tile_dim2=0
    g1[5] = CD;                                            // tensor_dim0_stride[31:0]=64
    g1[6] = 0;                                             // stride hi | dim1_stride lo
    g1[7] = 0;
    i32x4 g2 = {0, 0, 0, 0}, g3 = {0, 0, 0, 0};            // unused (2-D tensor)
    i32x8 g4 = {0, 0, 0, 0, 0, 0, 0, 0};                   // unused trailing group
    __builtin_amdgcn_tensor_load_to_lds(g0, g1, g2, g3, g4, 0);
}

__global__ __launch_bounds__(256) void fa_block_diag(
    const float* __restrict__ Q, const float* __restrict__ KV,
    const int* __restrict__ qcoor, const int* __restrict__ kcoor,
    float* __restrict__ O, int n1, int n2)
{
    __shared__ float  stageF[2][KT][CD];     // TDM fp32 staging, double-buffered
    __shared__ __bf16 Ks[KT][KPITCH];        // K tile, row-major (key, chan)
    __shared__ __bf16 Vt[CD][VPITCH];        // V tile, transposed (chan, key)
    __shared__ __bf16 Ps[WAVES][16][PPITCH]; // per-wave P tile (row, key)
    __shared__ int    kcs[KT];
    __shared__ int    qcs[BQ];

    const int tid  = threadIdx.x;
    const int lane = tid & 31;
    const int wave = tid >> 5;
    const int ln   = lane & 15;
    const int hi   = lane >> 4;
    const int qblk = blockIdx.x * BQ;
    const int qbase = qblk + wave * 16;

    // ---- query batch ids for this block ----
    if (tid < BQ) {
        int r = qblk + tid;
        qcs[tid] = qcoor[(r < n1) ? r : (n1 - 1)];
    }
    __syncthreads();

    int qcl[8];
#pragma unroll
    for (int v = 0; v < 8; ++v) qcl[v] = qcs[wave * 16 + v + 8 * hi];

    // ---- load Q 16x64 as two 16x32 bf16 A-fragments, 1/sqrt(d) folded in ----
    v16bf aq[2];
    {
        int r = qbase + ln; if (r >= n1) r = n1 - 1;
        const float* qrow = Q + (size_t)r * CD;
#pragma unroll
        for (int f = 0; f < 2; ++f) {
            const int cb = 32 * f + 8 * hi;
#pragma unroll
            for (int e = 0; e < 8; ++e) {
                aq[f][e]     = (__bf16)(qrow[cb + e]      * 0.125f);
                aq[f][8 + e] = (__bf16)(qrow[cb + 16 + e] * 0.125f);
            }
        }
    }

    // ---- all-ones B fragment: PV-style wmma with it yields row sums of P ----
    v16bf ones;
#pragma unroll
    for (int e = 0; e < 16; ++e) ones[e] = (__bf16)1.0f;

    // ---- key range for this block (coors are sorted) ----
    int blo = qcs[0], bhi = qcs[BQ - 1];
    if (bhi < blo) bhi = blo;
    int k_begin, k_end;
    {
        int lo = 0, up = n2;
        while (lo < up) { int mid = (lo + up) >> 1; if (kcoor[mid] <  blo) lo = mid + 1; else up = mid; }
        k_begin = lo & ~(KT - 1);
        lo = 0; up = n2;
        while (lo < up) { int mid = (lo + up) >> 1; if (kcoor[mid] <= bhi) lo = mid + 1; else up = mid; }
        k_end = lo;
    }

    // ---- softmax max state + output/normalizer accumulators ----
    float m_i[8];
    v8f oacc[4], lacc;
#pragma unroll
    for (int v = 0; v < 8; ++v) m_i[v] = -1e30f;
#pragma unroll
    for (int e = 0; e < 8; ++e) lacc[e] = 0.f;
#pragma unroll
    for (int ct = 0; ct < 4; ++ct)
#pragma unroll
        for (int e = 0; e < 8; ++e) oacc[ct][e] = 0.f;

    const unsigned stage_off0 = (unsigned)(size_t)&stageF[0][0][0]; // LDS byte offset
    const unsigned stage_off1 = (unsigned)(size_t)&stageF[1][0][0];

    // ---- prologue: TDM first tile into stage[0]; prefetch first kc ----
    if (wave == 0 && k_begin < k_end)
        tdm_load_tile(KV + (size_t)k_begin * CD, stage_off0, (unsigned)(n2 - k_begin));
    const int ld_key = tid >> 3;          // 0..31
    const int ld_cb  = (tid & 7) * 8;     // 0..56
    int nkc = -9;
    if (tid < KT && k_begin < k_end && k_begin + tid < n2) nkc = kcoor[k_begin + tid];
    if (wave == 0) __builtin_amdgcn_s_wait_tensorcnt(0);
    __syncthreads();                      // stage[0] visible to all waves

    int buf = 0;
    for (int kt = k_begin; kt < k_end; kt += KT) {
        // ---- kick off TDM for the next tile (overlaps convert + compute) ----
        if (wave == 0 && kt + KT < k_end)
            tdm_load_tile(KV + (size_t)(kt + KT) * CD,
                          buf ? stage_off0 : stage_off1,
                          (unsigned)(n2 - (kt + KT)));

        // ---- convert staged fp32 tile -> bf16 LDS tiles ----
        {
            const float* srow = &stageF[buf][ld_key][ld_cb];
            float4 f0 = *(const float4*)(srow);
            float4 f1 = *(const float4*)(srow + 4);
            v8bf pk;
            pk[0]=(__bf16)f0.x; pk[1]=(__bf16)f0.y; pk[2]=(__bf16)f0.z; pk[3]=(__bf16)f0.w;
            pk[4]=(__bf16)f1.x; pk[5]=(__bf16)f1.y; pk[6]=(__bf16)f1.z; pk[7]=(__bf16)f1.w;
            *(v8bf*)&Ks[ld_key][ld_cb] = pk;                            // ds_store_b128
#pragma unroll
            for (int e = 0; e < 8; ++e) Vt[ld_cb + e][ld_key] = pk[e];  // transposed
            if (tid < KT) kcs[tid] = nkc;
        }
        __syncthreads();

        // ---- prefetch next tile's key coors into a register ----
        {
            const int nkt = kt + KT;
            nkc = -9;
            if (tid < KT && nkt < k_end && nkt + tid < n2) nkc = kcoor[nkt + tid];
        }

        // ---- S = (Q/sqrt(d)) K^T : two 16x16 n-tiles, K-dim 64 = 2 chained wmma ----
        v8f s[2];
#pragma unroll
        for (int nt = 0; nt < 2; ++nt) {
            v8f c;
#pragma unroll
            for (int e = 0; e < 8; ++e) c[e] = 0.f;
#pragma unroll
            for (int f = 0; f < 2; ++f) {
                v16bf b;
                const __bf16* src = &Ks[nt * 16 + ln][32 * f + 8 * hi];
#pragma unroll
                for (int e = 0; e < 16; ++e) b[e] = src[(e < 8) ? e : (8 + e)];
                c = __builtin_amdgcn_wmma_f32_16x16x32_bf16(
                        false, aq[f], false, b, (short)0, c, false, false);
            }
            s[nt] = c;
        }

        // ---- mask + online softmax: DPP row-max, no LDS shuffles ----
        const int kc0 = kcs[ln], kc1 = kcs[16 + ln];
#pragma unroll
        for (int v = 0; v < 8; ++v) {
            const float s0 = (qcl[v] == kc0) ? s[0][v] : -1e30f;
            const float s1 = (qcl[v] == kc1) ? s[1][v] : -1e30f;
            const float t     = rowmax16(fmaxf(s0, s1));
            const float mnew  = fmaxf(m_i[v], t);
            const float alpha = __expf(m_i[v] - mnew);
            m_i[v] = mnew;
            const float p0 = __expf(s0 - mnew);
            const float p1 = __expf(s1 - mnew);
            Ps[wave][v + 8 * hi][ln]      = (__bf16)p0;
            Ps[wave][v + 8 * hi][16 + ln] = (__bf16)p1;
            lacc[v] *= alpha;
#pragma unroll
            for (int ct = 0; ct < 4; ++ct) oacc[ct][v] *= alpha;
        }
        __builtin_amdgcn_wave_barrier();   // keep Ps stores before Ps reads

        // ---- re-layout P into an A-fragment (16x32 bf16) ----
        v16bf ap;
        {
            const __bf16* p0 = &Ps[wave][ln][8 * hi];
            const __bf16* p1 = &Ps[wave][ln][16 + 8 * hi];
#pragma unroll
            for (int e = 0; e < 8; ++e) { ap[e] = p0[e]; ap[8 + e] = p1[e]; }
        }

        // ---- O += P V (4 channel tiles), l += P 1 (ones fragment) ----
#pragma unroll
        for (int ct = 0; ct < 4; ++ct) {
            v16bf bv;
            const __bf16* vr = &Vt[ct * 16 + ln][8 * hi];
#pragma unroll
            for (int e = 0; e < 16; ++e) bv[e] = vr[(e < 8) ? e : (8 + e)];
            oacc[ct] = __builtin_amdgcn_wmma_f32_16x16x32_bf16(
                           false, ap, false, bv, (short)0, oacc[ct], false, false);
        }
        lacc = __builtin_amdgcn_wmma_f32_16x16x32_bf16(
                   false, ap, false, ones, (short)0, lacc, false, false);

        // ---- ensure next tile's DMA has landed, then publish to all waves ----
        if (wave == 0) __builtin_amdgcn_s_wait_tensorcnt(0);
        __syncthreads();
        buf ^= 1;
    }

    // ---- epilogue: normalize and store (row sum already in every lane) ----
#pragma unroll
    for (int v = 0; v < 8; ++v) {
        const int r = qbase + v + 8 * hi;
        if (r < n1) {
            const float linv = (lacc[v] > 0.f) ? 1.f / lacc[v] : 0.f;
#pragma unroll
            for (int ct = 0; ct < 4; ++ct)
                O[(size_t)r * CD + ct * 16 + ln] = oacc[ct][v] * linv;
        }
    }
}

extern "C" void kernel_launch(void* const* d_in, const int* in_sizes, int n_in,
                              void* d_out, int out_size, void* d_ws, size_t ws_size,
                              hipStream_t stream) {
    const float* Q  = (const float*)d_in[0];
    const float* KV = (const float*)d_in[1];
    const int*   qc = (const int*)d_in[2];
    const int*   kc = (const int*)d_in[3];
    float*       O  = (float*)d_out;
    const int n1 = in_sizes[0] / CD;
    const int n2 = in_sizes[1] / CD;
    const int blocks = (n1 + BQ - 1) / BQ;
    fa_block_diag<<<blocks, 256, 0, stream>>>(Q, KV, qc, kc, O, n1, n2);
}
